// MoEModelBatched_20675972563214
// MI455X (gfx1250) — matrix operand order repaired and checked by reading, not compiled, not run on hardware
//
#include <hip/hip_runtime.h>

// Problem dims (match reference)
#define B_TOK 65536
#define D_IN  128
#define RHDIM 128
#define H1DIM 256
#define H2DIM 128
#define C_OUT 64
#define E_NUM 8

// Padded LDS row strides (in bf16 elements): +8 keeps 16B alignment for b128
// loads while rotating rows across banks (row*68 dwords -> +4 banks per row),
// making A-fragment ds_load_b128 conflict-free.
#define XS   136   // x / h2 tiles (width 128)
#define H1S  264   // h1 tile (width 256)
#define RH2S 72    // router h2 tile (width 64)

// Expert kernel tile: 128 tokens, 8 waves. Dynamic LDS partition:
#define ETOK 128
#define E_LDS_X   ((size_t)ETOK * XS * 2)
#define E_LDS_H1  ((size_t)ETOK * H1S * 2)
#define E_LDS_TOT (E_LDS_X + E_LDS_H1 + ETOK * 4 + ETOK * 4)

typedef unsigned short ushort_t;
typedef __attribute__((ext_vector_type(16))) __bf16 v16bf;
typedef __attribute__((ext_vector_type(8)))  float  v8f;

union FragU { v16bf v; uint4 q[2]; };

__device__ __forceinline__ ushort_t f2bf(float f) {
  unsigned u = __float_as_uint(f);
  u += 0x7FFFu + ((u >> 16) & 1u);   // round-to-nearest-even
  return (ushort_t)(u >> 16);
}

// A-matrix 16x32 bf16 fragment (row fixed per lane; K = kb..kb+7 and kb+16..kb+23)
__device__ __forceinline__ v16bf load_a_frag(const ushort_t* rowp, int k0, int lane) {
  int kb = k0 + ((lane >> 4) << 3);
  FragU f;
  f.q[0] = *(const uint4*)(rowp + kb);
  f.q[1] = *(const uint4*)(rowp + kb + 16);
  return f.v;
}

// B-matrix 32x16 bf16 fragment from output-major weight (col fixed per lane; K = kb..kb+7 and kb+16..kb+23)
__device__ __forceinline__ v16bf load_b_frag(const ushort_t* colp, int k0, int lane) {
  int kb = k0 + ((lane >> 4) << 3);
  FragU f;
  f.q[0] = *(const uint4*)(colp + kb);
  f.q[1] = *(const uint4*)(colp + kb + 16);
  return f.v;
}

#define WMMA_BF16(a, b, c) \
  __builtin_amdgcn_wmma_f32_16x16x32_bf16(false, (a), false, (b), (short)0, (c), false, false)

// ---------------------------------------------------------------------------
// Init: zero output accumulator (vectorized), scalar losses, expert counters
// ---------------------------------------------------------------------------
__global__ void k_init(float4* __restrict__ out_y4, float* __restrict__ scalars,
                       int* __restrict__ counts) {
  size_t i = (size_t)blockIdx.x * 256 + threadIdx.x;
  if (i < (size_t)B_TOK * C_OUT / 4) out_y4[i] = make_float4(0.f, 0.f, 0.f, 0.f);
  if (blockIdx.x == 0) {
    if (threadIdx.x < 2) scalars[threadIdx.x] = 0.0f;
    if (threadIdx.x >= 32 && threadIdx.x < 32 + E_NUM) counts[threadIdx.x - 32] = 0;
  }
}

// fp32 -> bf16 bulk convert (for x)
__global__ void k_cvt(const float* __restrict__ in, ushort_t* __restrict__ out, size_t n) {
  size_t i = (size_t)blockIdx.x * 256 + threadIdx.x;
  if (i < n) out[i] = f2bf(in[i]);
}

// fp32 [batch][R][C] -> bf16 [batch][C][R]  (weights become output-major)
__global__ void k_transpose_bf16(const float* __restrict__ in, ushort_t* __restrict__ out,
                                 int R, int C) {
  const float* src = in + (size_t)blockIdx.y * R * C;
  ushort_t* dst = out + (size_t)blockIdx.y * R * C;
  int n = R * C;
  for (int i = blockIdx.x * 256 + threadIdx.x; i < n; i += gridDim.x * 256) {
    int r = i / C, c = i % C;
    dst[c * R + r] = f2bf(src[i]);
  }
}

// rW3 [64][8] fp32 -> [16][64] bf16 padded with zeros (cols 8..15)
__global__ void k_rw3t(const float* __restrict__ rW3, ushort_t* __restrict__ out) {
  int t = threadIdx.x;           // 1024 threads = 16*64
  int n = t >> 6, k = t & 63;
  out[n * 64 + k] = (n < 8) ? f2bf(rW3[k * 8 + n]) : (ushort_t)0;
}

// ---------------------------------------------------------------------------
// Router: 64-token tile, 4 waves, WMMA bf16 chain 128->128->64->8(+pad),
// softmax, top-2, block-aggregated dispatch, block-reduced diversity loss.
// ---------------------------------------------------------------------------
__global__ __launch_bounds__(128)
void k_router(const ushort_t* __restrict__ xb,
              const ushort_t* __restrict__ w1t, const float* __restrict__ b1,
              const ushort_t* __restrict__ w2t, const float* __restrict__ b2,
              const ushort_t* __restrict__ w3t, const float* __restrict__ b3,
              float* __restrict__ probs_out, float* __restrict__ combine,
              int* __restrict__ counts, int* __restrict__ lists,
              float* __restrict__ div_out) {
  __shared__ ushort_t ls_x[64 * XS];
  __shared__ ushort_t ls_h1[64 * XS];
  __shared__ ushort_t ls_h2[64 * RH2S];
  __shared__ float    ls_sc[64 * 16];
  __shared__ float    red[64];
  __shared__ int      l_cnt[E_NUM], l_base[E_NUM];

  const int t    = threadIdx.x;
  const int lane = t & 31;
  const int wave = t >> 5;
  const int tok0 = blockIdx.x * 64;

  if (t < E_NUM) l_cnt[t] = 0;

  { // load 64 x-rows (bf16), 2 threads per row
    int r = t >> 1, half = t & 1;
    const uint4* src = (const uint4*)(xb + (size_t)(tok0 + r) * D_IN + half * 64);
    uint4* dst = (uint4*)(ls_x + r * XS + half * 64);
#pragma unroll
    for (int i = 0; i < 8; ++i) dst[i] = src[i];
  }
  __syncthreads();

  const int mrow = wave * 16 + (lane & 15);
  const int ncol = lane & 15;
  const int rhi  = (lane >> 4) << 3;   // C/D row offset 0 or 8

  { // layer 1: [64,128] x [128,128] -> relu -> h1
    v16bf a[4];
#pragma unroll
    for (int ks = 0; ks < 4; ++ks) a[ks] = load_a_frag(ls_x + mrow * XS, ks * 32, lane);
    for (int nt = 0; nt < 8; ++nt) {
      v8f acc = {};
      int n = nt * 16 + ncol;
#pragma unroll
      for (int ks = 0; ks < 4; ++ks)
        acc = WMMA_BF16(a[ks], load_b_frag(w1t + (size_t)n * D_IN, ks * 32, lane), acc);
      float bias = b1[n];
#pragma unroll
      for (int r = 0; r < 8; ++r) {
        float v = acc[r] + bias;
        ls_h1[(wave * 16 + rhi + r) * XS + n] = f2bf(v > 0.f ? v : 0.f);
      }
    }
  }
  __syncthreads();

  { // layer 2: [64,128] x [128,64] -> relu -> h2
    v16bf a[4];
#pragma unroll
    for (int ks = 0; ks < 4; ++ks) a[ks] = load_a_frag(ls_h1 + mrow * XS, ks * 32, lane);
    for (int nt = 0; nt < 4; ++nt) {
      v8f acc = {};
      int n = nt * 16 + ncol;
#pragma unroll
      for (int ks = 0; ks < 4; ++ks)
        acc = WMMA_BF16(a[ks], load_b_frag(w2t + (size_t)n * RHDIM, ks * 32, lane), acc);
      float bias = b2[n];
#pragma unroll
      for (int r = 0; r < 8; ++r) {
        float v = acc[r] + bias;
        ls_h2[(wave * 16 + rhi + r) * RH2S + n] = f2bf(v > 0.f ? v : 0.f);
      }
    }
  }
  __syncthreads();

  { // layer 3: [64,64] x [64,16(pad)] -> raw scores
    v8f acc = {};
#pragma unroll
    for (int ks = 0; ks < 2; ++ks)
      acc = WMMA_BF16(load_a_frag(ls_h2 + mrow * RH2S, ks * 32, lane),
                      load_b_frag(w3t + (size_t)ncol * 64, ks * 32, lane), acc);
    float bias = (ncol < 8) ? b3[ncol] : 0.f;
#pragma unroll
    for (int r = 0; r < 8; ++r)
      ls_sc[(wave * 16 + rhi + r) * 16 + ncol] = acc[r] + bias;
  }
  __syncthreads();

  // per-token softmax / top-2; block-local dispatch aggregation
  int i1 = 0, i2 = 0, p1 = 0, p2 = 0, tok = tok0 + t;
  if (t < 64) {
    float s[8], m = -3.0e38f, dv = 0.f;
#pragma unroll
    for (int j = 0; j < 8; ++j) { s[j] = ls_sc[t * 16 + j]; dv += s[j] * s[j]; m = s[j] > m ? s[j] : m; }
    float sum = 0.f;
#pragma unroll
    for (int j = 0; j < 8; ++j) { s[j] = __expf(s[j] - m); sum += s[j]; }
    float inv = 1.0f / sum;
#pragma unroll
    for (int j = 0; j < 8; ++j) { s[j] *= inv; probs_out[(size_t)tok * 8 + j] = s[j]; }
#pragma unroll
    for (int j = 1; j < 8; ++j) if (s[j] > s[i1]) i1 = j;
    i2 = (i1 == 0) ? 1 : 0;
#pragma unroll
    for (int j = 0; j < 8; ++j) if (j != i1 && s[j] > s[i2]) i2 = j;
    combine[(size_t)tok * 8 + i1] = s[i1];
    combine[(size_t)tok * 8 + i2] = s[i2];
    p1 = atomicAdd(&l_cnt[i1], 1);
    p2 = atomicAdd(&l_cnt[i2], 1);
    red[t] = dv;
  }
  __syncthreads();
  if (t < E_NUM) l_base[t] = atomicAdd(&counts[t], l_cnt[t]);   // 8 global atomics/block
  if (t == 64) {                                                // 1 diversity atomic/block
    float s = 0.f;
    for (int j = 0; j < 64; ++j) s += red[j];
    atomicAdd(div_out, s);
  }
  __syncthreads();
  if (t < 64) {
    lists[i1 * B_TOK + l_base[i1] + p1] = tok;
    lists[i2 * B_TOK + l_base[i2] + p2] = tok;
  }
}

// ---------------------------------------------------------------------------
// Experts: 128-token sparse gather tile, 8 waves, bf16 WMMA chain
// 128->256->128->64, scaled atomic combine. Dynamic LDS (~101 KB/WG, legal on
// CDNA5's 320 KB WGP LDS). Blocks beyond this expert's count exit immediately.
// ---------------------------------------------------------------------------
__global__ __launch_bounds__(256)
void k_expert(const ushort_t* __restrict__ xb,
              const ushort_t* __restrict__ w1t_all, const float* __restrict__ b1_all,
              const ushort_t* __restrict__ w2t_all, const float* __restrict__ b2_all,
              const ushort_t* __restrict__ w3t_all, const float* __restrict__ b3_all,
              const int* __restrict__ counts, const int* __restrict__ lists,
              const float* __restrict__ combine,
              float* __restrict__ out) {
  extern __shared__ char smem[];
  ushort_t* ls_x  = (ushort_t*)smem;                 // [128][XS], reused as h2
  ushort_t* ls_h1 = ls_x + ETOK * XS;                // [128][H1S]
  int*      toks  = (int*)(ls_h1 + ETOK * H1S);
  float*    wts   = (float*)(toks + ETOK);

  const int e    = blockIdx.y;
  const int cnt  = counts[e];
  const int base = blockIdx.x * ETOK;
  if (base >= cnt) return;                 // device-side sparsity

  const ushort_t* w1 = w1t_all + (size_t)e * H1DIM * D_IN;
  const ushort_t* w2 = w2t_all + (size_t)e * H2DIM * H1DIM;
  const ushort_t* w3 = w3t_all + (size_t)e * C_OUT * H2DIM;
  const float* b1 = b1_all + e * H1DIM;
  const float* b2 = b2_all + e * H2DIM;
  const float* b3 = b3_all + e * C_OUT;

  const int t    = threadIdx.x;
  const int lane = t & 31;
  const int wave = t >> 5;

  if (t < ETOK) {
    int idx = base + t;
    int tok = (idx < cnt) ? lists[e * B_TOK + idx] : -1;
    toks[t] = tok;
    wts[t]  = (tok >= 0) ? combine[(size_t)tok * 8 + e] * 0.5f : 0.f;  // /top_k
  }
  __syncthreads();

  { // gather 128 x-rows, 2 threads per row
    int r = t >> 1, half = t & 1;
    int tok = toks[r];
    uint4* dst = (uint4*)(ls_x + r * XS + half * 64);
    if (tok >= 0) {
      const uint4* src = (const uint4*)(xb + (size_t)tok * D_IN + half * 64);
#pragma unroll
      for (int i = 0; i < 8; ++i) dst[i] = src[i];
    } else {
      uint4 z = make_uint4(0, 0, 0, 0);
#pragma unroll
      for (int i = 0; i < 8; ++i) dst[i] = z;
    }
  }
  __syncthreads();

  const int mrow = wave * 16 + (lane & 15);
  const int ncol = lane & 15;
  const int rhi  = (lane >> 4) << 3;

  { // layer 1: [128,128] x [128,256] -> relu -> h1
    v16bf a[4];
#pragma unroll
    for (int ks = 0; ks < 4; ++ks) a[ks] = load_a_frag(ls_x + mrow * XS, ks * 32, lane);
    for (int nt = 0; nt < 16; ++nt) {
      v8f acc = {};
      int n = nt * 16 + ncol;
#pragma unroll
      for (int ks = 0; ks < 4; ++ks)
        acc = WMMA_BF16(a[ks], load_b_frag(w1 + (size_t)n * D_IN, ks * 32, lane), acc);
      float bias = b1[n];
#pragma unroll
      for (int r = 0; r < 8; ++r) {
        float v = acc[r] + bias;
        ls_h1[(wave * 16 + rhi + r) * H1S + n] = f2bf(v > 0.f ? v : 0.f);
      }
    }
  }
  __syncthreads();

  { // layer 2: [128,256] x [256,128] -> relu -> h2 (into ls_x region)
    v16bf a[8];
#pragma unroll
    for (int ks = 0; ks < 8; ++ks) a[ks] = load_a_frag(ls_h1 + mrow * H1S, ks * 32, lane);
    for (int nt = 0; nt < 8; ++nt) {
      v8f acc = {};
      int n = nt * 16 + ncol;
#pragma unroll
      for (int ks = 0; ks < 8; ++ks)
        acc = WMMA_BF16(a[ks], load_b_frag(w2 + (size_t)n * H1DIM, ks * 32, lane), acc);
      float bias = b2[n];
#pragma unroll
      for (int r = 0; r < 8; ++r) {
        float v = acc[r] + bias;
        ls_x[(wave * 16 + rhi + r) * XS + n] = f2bf(v > 0.f ? v : 0.f);
      }
    }
  }
  __syncthreads();

  { // layer 3: [128,128] x [128,64] -> scale by combine/2 -> atomic add into out
    v16bf a[4];
#pragma unroll
    for (int ks = 0; ks < 4; ++ks) a[ks] = load_a_frag(ls_x + mrow * XS, ks * 32, lane);
    for (int nt = 0; nt < 4; ++nt) {
      v8f acc = {};
      int n = nt * 16 + ncol;
#pragma unroll
      for (int ks = 0; ks < 4; ++ks)
        acc = WMMA_BF16(a[ks], load_b_frag(w3 + (size_t)n * H2DIM, ks * 32, lane), acc);
      float bias = b3[n];
#pragma unroll
      for (int r = 0; r < 8; ++r) {
        int row = wave * 16 + rhi + r;
        int tok = toks[row];
        if (tok >= 0) {
          float v = (acc[r] + bias) * wts[row];
          atomicAdd(out + (size_t)tok * C_OUT + n, v);
        }
      }
    }
  }
}

// ---------------------------------------------------------------------------
extern "C" void kernel_launch(void* const* d_in, const int* in_sizes, int n_in,
                              void* d_out, int out_size, void* d_ws, size_t ws_size,
                              hipStream_t stream) {
  (void)in_sizes; (void)n_in; (void)out_size; (void)ws_size;
  const float* x   = (const float*)d_in[0];
  const float* rW1 = (const float*)d_in[1];
  const float* rb1 = (const float*)d_in[2];
  const float* rW2 = (const float*)d_in[3];
  const float* rb2 = (const float*)d_in[4];
  const float* rW3 = (const float*)d_in[5];
  const float* rb3 = (const float*)d_in[6];
  const float* eW1 = (const float*)d_in[7];
  const float* eb1 = (const float*)d_in[8];
  const float* eW2 = (const float*)d_in[9];
  const float* eb2 = (const float*)d_in[10];
  const float* eW3 = (const float*)d_in[11];
  const float* eb3 = (const float*)d_in[12];

  float* out_y  = (float*)d_out;                              // [B,64]
  float* probs  = out_y + (size_t)B_TOK * C_OUT;              // [B,8]
  float* scal   = probs + (size_t)B_TOK * E_NUM;              // lb_loss, diversity

  char* ws = (char*)d_ws;
  size_t off = 0;
  auto alloc = [&](size_t bytes) {
    char* p = ws + off;
    off = (off + bytes + 255) & ~(size_t)255;
    return p;
  };
  ushort_t* xb    = (ushort_t*)alloc((size_t)B_TOK * D_IN * 2);
  ushort_t* rw1t  = (ushort_t*)alloc((size_t)RHDIM * D_IN * 2);
  ushort_t* rw2t  = (ushort_t*)alloc((size_t)64 * RHDIM * 2);
  ushort_t* rw3t  = (ushort_t*)alloc((size_t)16 * 64 * 2);
  ushort_t* ew1t  = (ushort_t*)alloc((size_t)E_NUM * H1DIM * D_IN * 2);
  ushort_t* ew2t  = (ushort_t*)alloc((size_t)E_NUM * H2DIM * H1DIM * 2);
  ushort_t* ew3t  = (ushort_t*)alloc((size_t)E_NUM * C_OUT * H2DIM * 2);
  float*    comb  = (float*)alloc((size_t)B_TOK * E_NUM * 4);
  int*      cnts  = (int*)alloc(E_NUM * 4);
  int*      lsts  = (int*)alloc((size_t)E_NUM * B_TOK * 4);

  // 1. init (vectorized zero of output accumulator)
  k_init<<<dim3(((size_t)B_TOK * C_OUT / 4 + 255) / 256), dim3(256), 0, stream>>>(
      (float4*)out_y, scal, cnts);
  // 2. convert x -> bf16
  k_cvt<<<dim3(((size_t)B_TOK * D_IN + 255) / 256), dim3(256), 0, stream>>>(
      x, xb, (size_t)B_TOK * D_IN);
  // 3. weights -> bf16, output-major
  k_transpose_bf16<<<dim3(64, 1), dim3(256), 0, stream>>>(rW1, rw1t, D_IN, RHDIM);
  k_transpose_bf16<<<dim3(32, 1), dim3(256), 0, stream>>>(rW2, rw2t, RHDIM, 64);
  k_rw3t<<<dim3(1), dim3(1024), 0, stream>>>(rW3, rw3t);
  k_transpose_bf16<<<dim3(128, E_NUM), dim3(256), 0, stream>>>(eW1, ew1t, D_IN, H1DIM);
  k_transpose_bf16<<<dim3(128, E_NUM), dim3(256), 0, stream>>>(eW2, ew2t, H1DIM, H2DIM);
  k_transpose_bf16<<<dim3(32,  E_NUM), dim3(256), 0, stream>>>(eW3, ew3t, H2DIM, C_OUT);
  // 4. router + dispatch
  k_router<<<dim3(B_TOK / 64), dim3(128), 0, stream>>>(xb, rw1t, rb1, rw2t, rb2, rw3t, rb3,
                                                       probs, comb, cnts, lsts, scal + 1);
  // 5. sparse expert GEMMs + combine (dynamic LDS ~101 KB)
  k_expert<<<dim3(B_TOK / ETOK, E_NUM), dim3(256), (unsigned)E_LDS_TOT, stream>>>(
      xb, ew1t, eb1, ew2t, eb2, ew3t, eb3, cnts, lsts, comb, out_y);
}
